// RelativeAttentionLayer_2628519985846
// MI455X (gfx1250) — matrix operand order repaired and checked by reading, not compiled, not run on hardware
//
#include <hip/hip_runtime.h>
#include <hip/hip_bf16.h>

// ---------------------------------------------------------------------------
// Problem constants (fixed by the reference)
// ---------------------------------------------------------------------------
#define DM 1024      // d_model
#define NHD 16       // heads
#define HD 64        // head dim
#define LQ 2048      // query len
#define SK 2048      // key len
#define NB 2         // batch
#define MROWS (NB * LQ)   // 4096 rows for the projection GEMMs

typedef __attribute__((ext_vector_type(16))) _Float16 v16h;
typedef __attribute__((ext_vector_type(8)))  _Float16 v8h;
typedef __attribute__((ext_vector_type(8)))  float    v8f;

// ---------------------------------------------------------------------------
// WMMA helpers (CDNA5 gfx1250, wave32, 16x16x32 f16 -> f32)
// ---------------------------------------------------------------------------
__device__ __forceinline__ v8f wmma_f16(v16h a, v16h b, v8f c) {
  return __builtin_amdgcn_wmma_f32_16x16x32_f16(false, a, false, b, (short)0, c,
                                                false, false);
}

// A-matrix fragment (16x32, row-major, ld in halfs).
__device__ __forceinline__ v16h load_a_frag(const _Float16* __restrict__ a, int lda) {
  int lane = threadIdx.x & 31;
  int row  = lane & 15;
  int kb   = (lane >> 4) << 4;
  const _Float16* p = a + (size_t)row * lda + kb;
  v8h lo = *(const v8h*)p;
  v8h hi = *(const v8h*)(p + 8);
  v16h r;
#pragma unroll
  for (int i = 0; i < 8; ++i) { r[i] = lo[i]; r[i + 8] = hi[i]; }
  return r;
}

// B-matrix fragment (32x16, B[k][n] = src[n][k], ld in halfs).
__device__ __forceinline__ v16h load_b_frag(const _Float16* __restrict__ w, int ldw) {
  int lane = threadIdx.x & 31;
  const _Float16* p = w + (size_t)(lane & 15) * ldw + ((lane >> 4) << 4);
  return *(const v16h*)p;
}

// Same B-fragment pattern out of an LDS tile (64 rows x 64 halfs, row-major).
__device__ __forceinline__ v16h load_b_frag_lds(const _Float16* tile, int row0, int koff) {
  int lane = threadIdx.x & 31;
  const _Float16* p = tile + (size_t)(row0 + (lane & 15)) * 64 +
                      ((lane >> 4) << 4) + koff;
  return *(const v16h*)p;
}

// ---------------------------------------------------------------------------
// CDNA5 async global->LDS copy (ASYNCcnt-tracked).  INST_OFFSET is added to
// BOTH the LDS and global address, so a linear 8KB tile copy needs only one
// (lds,global) base pair and 16 immediate offsets -> 3 live VGPRs total.
// ---------------------------------------------------------------------------
#define ASYNC_LD_OFF(OFF)                                                  \
  asm volatile("global_load_async_to_lds_b128 %0, %1, off offset:" #OFF    \
               :: "v"(base), "v"(g) : "memory")

__device__ __forceinline__ void stage_k_tile(_Float16* lds_tile, const _Float16* kg) {
  int lane = threadIdx.x & 31;
  unsigned base = (unsigned)(size_t)lds_tile + (unsigned)lane * 16u;
  const char* g = (const char*)kg + (size_t)lane * 16;
  ASYNC_LD_OFF(0);    ASYNC_LD_OFF(512);  ASYNC_LD_OFF(1024); ASYNC_LD_OFF(1536);
  ASYNC_LD_OFF(2048); ASYNC_LD_OFF(2560); ASYNC_LD_OFF(3072); ASYNC_LD_OFF(3584);
  ASYNC_LD_OFF(4096); ASYNC_LD_OFF(4608); ASYNC_LD_OFF(5120); ASYNC_LD_OFF(5632);
  ASYNC_LD_OFF(6144); ASYNC_LD_OFF(6656); ASYNC_LD_OFF(7168); ASYNC_LD_OFF(7680);
}

// ---------------------------------------------------------------------------
// fp32 -> fp16 conversion
// ---------------------------------------------------------------------------
__global__ void cvt_f32_f16(const float* __restrict__ x, _Float16* __restrict__ y,
                            size_t n) {
  size_t i = (size_t)blockIdx.x * blockDim.x + threadIdx.x;
  if (i < n) y[i] = (_Float16)x[i];
}

// ---------------------------------------------------------------------------
// GEMM: C[M x N] = A[M x K] * W^T (+bias), W stored [N x K] row-major.
// 8 waves/block, 64x64 C tile per wave, block tile 128 x 256.  Fed straight
// from global: the whole 85MB working set lives in the 192MB L2.
// ---------------------------------------------------------------------------
template <bool STORE_F16>
__global__ __launch_bounds__(256) void gemm_xwT_bias(
    const _Float16* __restrict__ A, const _Float16* __restrict__ W,
    const float* __restrict__ bias, void* __restrict__ Cout, int M, int Nn, int K) {
  int w    = threadIdx.x >> 5;
  int lane = threadIdx.x & 31;
  int m0 = blockIdx.x * 128 + (w & 1) * 64;
  int n0 = blockIdx.y * 256 + (w >> 1) * 64;

  v8f acc[4][4];
#pragma unroll
  for (int i = 0; i < 4; ++i)
#pragma unroll
    for (int j = 0; j < 4; ++j)
#pragma unroll
      for (int r = 0; r < 8; ++r) acc[i][j][r] = 0.f;

  for (int k0 = 0; k0 < K; k0 += 32) {
    v16h a[4], b[4];
#pragma unroll
    for (int i = 0; i < 4; ++i)
      a[i] = load_a_frag(A + (size_t)(m0 + 16 * i) * K + k0, K);
#pragma unroll
    for (int j = 0; j < 4; ++j)
      b[j] = load_b_frag(W + (size_t)(n0 + 16 * j) * K + k0, K);
#pragma unroll
    for (int i = 0; i < 4; ++i)
#pragma unroll
      for (int j = 0; j < 4; ++j) acc[i][j] = wmma_f16(a[i], b[j], acc[i][j]);
  }

  int ncol = lane & 15, mbase = (lane >> 4) * 8;
#pragma unroll
  for (int j = 0; j < 4; ++j) {
    int n = n0 + j * 16 + ncol;
    float bv = bias[n];
#pragma unroll
    for (int i = 0; i < 4; ++i) {
#pragma unroll
      for (int r = 0; r < 8; ++r) {
        float vv = acc[i][j][r] + bv;
        size_t off = (size_t)(m0 + i * 16 + mbase + r) * Nn + n;
        if constexpr (STORE_F16) ((_Float16*)Cout)[off] = (_Float16)vv;
        else                     ((float*)Cout)[off]    = vv;
      }
    }
  }
}

// ---------------------------------------------------------------------------
// RoPE + reshape: P (N,L,DM) f16 -> R (N,H,L,HD) f16, with an output scale so
// the attention kernel's 1/sqrt(hd) folds into Q for free.
// ---------------------------------------------------------------------------
__global__ void rope_reshape(const _Float16* __restrict__ P, _Float16* __restrict__ R,
                             float scale) {
  size_t idx = (size_t)blockIdx.x * blockDim.x + threadIdx.x;
  size_t total = (size_t)NB * LQ * NHD * 32;
  if (idx >= total) return;
  int d = (int)(idx & 31);
  int h = (int)((idx >> 5) & 15);
  size_t nl = idx >> 9;
  int l = (int)(nl & (LQ - 1));
  int n = (int)(nl >> 11);
  const _Float16* src = P + nl * DM + (size_t)h * HD;
  float x1 = (float)src[d];
  float x2 = (float)src[d + 32];
  float inv = __powf(10000.f, -(float)d * (1.f / 32.f));
  float th = (float)l * inv;
  float c = __cosf(th), s = __sinf(th);
  _Float16* dst = R + ((size_t)(n * NHD + h) * LQ + l) * HD;
  dst[d]      = (_Float16)((x1 * c - x2 * s) * scale);
  dst[d + 32] = (_Float16)((x2 * c + x1 * s) * scale);
}

// V reshape: P (N,S,DM) -> T (N,H,HD,S) so PV's B-fragments are contiguous.
__global__ void v_reshape(const _Float16* __restrict__ P, _Float16* __restrict__ T) {
  size_t idx = (size_t)blockIdx.x * blockDim.x + threadIdx.x;
  size_t total = (size_t)NB * SK * DM;
  if (idx >= total) return;
  int e = (int)(idx & (DM - 1));
  size_t ns = idx >> 10;
  int s = (int)(ns & (SK - 1));
  int n = (int)(ns >> 11);
  int h = e >> 6, d = e & 63;
  T[((size_t)((n * NHD + h) * HD + d)) * SK + s] = P[idx];
}

// ---------------------------------------------------------------------------
// Flash attention (causal, online softmax).  grid = (N*H, L/128); 8 waves per
// block, each wave owns 16 query rows and loops only to its own diagonal
// (s0 < ql0+16), applying the mask only on diagonal-straddling tiles.
// K tiles stream into per-wave double-buffered LDS via async b128 copies.
// `sc` is kept read-only after the WMMAs (mask recomputed in the exp pass)
// to avoid insert-chain register pressure; waves_per_eu=2 curbs pipelining.
// ---------------------------------------------------------------------------
__global__ __launch_bounds__(256, 2) void flash_attn(
    const _Float16* __restrict__ Q,   // (N,H,L,HD), pre-scaled by 1/8
    const _Float16* __restrict__ Kx,  // (N,H,S,HD)
    const _Float16* __restrict__ Vt,  // (N,H,HD,S)
    _Float16* __restrict__ Ao) {      // (N,L,DM)
  __shared__ __align__(16) _Float16 kbuf[8][2][64 * 64];  // 128KB K staging
  __shared__ __align__(16) _Float16 lds_p[8][16][80];     // 20KB  P relayout
  int w    = threadIdx.x >> 5;
  int lane = threadIdx.x & 31;
  int nh = blockIdx.x;
  int n = nh >> 4, h = nh & 15;
  const _Float16* q = Q  + (size_t)nh * LQ * HD;
  const _Float16* k = Kx + (size_t)nh * SK * HD;
  const _Float16* v = Vt + (size_t)nh * HD * SK;
  int ql0 = blockIdx.y * 128 + w * 16;

  v16h qa0 = load_a_frag(q + (size_t)ql0 * HD, HD);
  v16h qa1 = load_a_frag(q + (size_t)ql0 * HD + 32, HD);

  v8f o[4];
  float mrow[8], lrow[8];
#pragma unroll
  for (int c = 0; c < 4; ++c)
#pragma unroll
    for (int r = 0; r < 8; ++r) o[c][r] = 0.f;
#pragma unroll
  for (int r = 0; r < 8; ++r) { mrow[r] = -1e30f; lrow[r] = 0.f; }

  int ncol = lane & 15, mbase = (lane >> 4) * 8;
  int s_last = ql0 + 16;  // exclusive; true per-wave causal bound

  stage_k_tile(&kbuf[w][0][0], k);  // tile 0 in flight

  for (int s0 = 0; s0 < s_last; s0 += 64) {
    int cur = (s0 >> 6) & 1;
    // WAR: all DS reads of the alternate buffer (previous iteration) retired.
    asm volatile("s_wait_dscnt 0" ::: "memory");
    // Prefetch next tile unconditionally (stays inside ws; unused on last).
    stage_k_tile(&kbuf[w][cur ^ 1][0], k + (size_t)(s0 + 64) * HD);
    // RAW: the 16 newest async ops are the prefetch; <=16 left => tile resident.
    asm volatile("s_wait_asynccnt 16" ::: "memory");
    const _Float16* ktile = &kbuf[w][cur][0];

    // ---- scores S = (Q/8) K^T, 16 x 64 per wave ----
    v8f sc[4];
#pragma unroll
    for (int j = 0; j < 4; ++j) {
#pragma unroll
      for (int r = 0; r < 8; ++r) sc[j][r] = 0.f;
      sc[j] = wmma_f16(qa0, load_b_frag_lds(ktile, j * 16, 0),  sc[j]);
      sc[j] = wmma_f16(qa1, load_b_frag_lds(ktile, j * 16, 32), sc[j]);
    }

    bool diag = (s0 + 63 > ql0);  // only diagonal-straddling tiles need a mask

    // ---- online softmax; sc stays read-only, mask recomputed per pass ----
#pragma unroll
    for (int r = 0; r < 8; ++r) {
      int lg = ql0 + mbase + r;
      float mx = -1e30f;
#pragma unroll
      for (int j = 0; j < 4; ++j) {
        float x = sc[j][r];
        bool valid = !diag || (s0 + j * 16 + ncol <= lg);
        mx = fmaxf(mx, valid ? x : -1e30f);
      }
      mx = fmaxf(mx, __shfl_xor(mx, 1, 32));
      mx = fmaxf(mx, __shfl_xor(mx, 2, 32));
      mx = fmaxf(mx, __shfl_xor(mx, 4, 32));
      mx = fmaxf(mx, __shfl_xor(mx, 8, 32));
      float mnew  = fmaxf(mrow[r], mx);
      float alpha = __expf(mrow[r] - mnew);
      mrow[r] = mnew;
      float rs = 0.f;
#pragma unroll
      for (int j = 0; j < 4; ++j) {
        bool valid = !diag || (s0 + j * 16 + ncol <= lg);
        float p = valid ? __expf(sc[j][r] - mnew) : 0.f;
        rs += p;
        lds_p[w][mbase + r][j * 16 + ncol] = (_Float16)p;
      }
      rs += __shfl_xor(rs, 1, 32);
      rs += __shfl_xor(rs, 2, 32);
      rs += __shfl_xor(rs, 4, 32);
      rs += __shfl_xor(rs, 8, 32);
      lrow[r] = lrow[r] * alpha + rs;
#pragma unroll
      for (int c = 0; c < 4; ++c) o[c][r] *= alpha;
    }

    // In-wave LDS RAW for the scattered P stores.
    asm volatile("s_wait_dscnt 0" ::: "memory");

    // ---- O += P V ----
#pragma unroll
    for (int kk = 0; kk < 2; ++kk) {
      v16h pa;
      {
        int row = lane & 15;
        int kb  = ((lane >> 4) << 4) + kk * 32;
        v8h lo = *(const v8h*)&lds_p[w][row][kb];
        v8h hi = *(const v8h*)&lds_p[w][row][kb + 8];
#pragma unroll
        for (int i = 0; i < 8; ++i) { pa[i] = lo[i]; pa[i + 8] = hi[i]; }
      }
#pragma unroll
      for (int c = 0; c < 4; ++c) {
        v16h vb = load_b_frag(v + (size_t)(c * 16) * SK + s0 + kk * 32, SK);
        o[c] = wmma_f16(pa, vb, o[c]);
      }
    }
  }

  // drain our speculative prefetch before the wave exits the LDS buffers
  asm volatile("s_wait_asynccnt 0" ::: "memory");

#pragma unroll
  for (int c = 0; c < 4; ++c)
#pragma unroll
    for (int r = 0; r < 8; ++r) {
      int lg = ql0 + mbase + r;
      size_t off = ((size_t)n * LQ + lg) * DM + (size_t)h * HD + c * 16 + ncol;
      Ao[off] = (_Float16)(o[c][r] / lrow[r]);
    }
}

// ---------------------------------------------------------------------------
// Host orchestration
// ---------------------------------------------------------------------------
extern "C" void kernel_launch(void* const* d_in, const int* in_sizes, int n_in,
                              void* d_out, int out_size, void* d_ws, size_t ws_size,
                              hipStream_t stream) {
  (void)in_sizes; (void)n_in; (void)out_size; (void)ws_size;
  const float* queries = (const float*)d_in[0];
  const float* keys    = (const float*)d_in[1];
  const float* values  = (const float*)d_in[2];
  const float* Wq = (const float*)d_in[3];  const float* bq = (const float*)d_in[4];
  const float* Wk = (const float*)d_in[5];  const float* bk = (const float*)d_in[6];
  const float* Wv = (const float*)d_in[7];  const float* bv = (const float*)d_in[8];
  const float* Wo = (const float*)d_in[9];  const float* bo = (const float*)d_in[10];
  float* out = (float*)d_out;

  const size_t NLD = (size_t)NB * LQ * DM;  // 4Mi elems
  const size_t DD  = (size_t)DM * DM;       // 1Mi elems
  _Float16* ws   = (_Float16*)d_ws;         // 28Mi f16 = 56MB used
  _Float16* X16Q = ws;
  _Float16* X16K = X16Q + NLD;
  _Float16* X16V = X16K + NLD;
  _Float16* W16Q = X16V + NLD;
  _Float16* W16K = W16Q + DD;
  _Float16* W16V = W16K + DD;
  _Float16* W16O = W16V + DD;
  _Float16* QP   = W16O + DD;
  _Float16* KP   = QP + NLD;
  _Float16* VP   = KP + NLD;
  _Float16* QR = X16Q;  // rope(QP), scaled by 1/8
  _Float16* KR = X16K;  // rope(KP)
  _Float16* VT = X16V;  // transpose(VP)
  _Float16* AO = QP;    // attention output

  auto nblk = [](size_t n) { return (unsigned)((n + 255) / 256); };

  cvt_f32_f16<<<nblk(NLD), 256, 0, stream>>>(queries, X16Q, NLD);
  cvt_f32_f16<<<nblk(NLD), 256, 0, stream>>>(keys,    X16K, NLD);
  cvt_f32_f16<<<nblk(NLD), 256, 0, stream>>>(values,  X16V, NLD);
  cvt_f32_f16<<<nblk(DD),  256, 0, stream>>>(Wq, W16Q, DD);
  cvt_f32_f16<<<nblk(DD),  256, 0, stream>>>(Wk, W16K, DD);
  cvt_f32_f16<<<nblk(DD),  256, 0, stream>>>(Wv, W16V, DD);
  cvt_f32_f16<<<nblk(DD),  256, 0, stream>>>(Wo, W16O, DD);

  dim3 gg(MROWS / 128, DM / 256);
  gemm_xwT_bias<true><<<gg, 256, 0, stream>>>(X16Q, W16Q, bq, QP, MROWS, DM, DM);
  gemm_xwT_bias<true><<<gg, 256, 0, stream>>>(X16K, W16K, bk, KP, MROWS, DM, DM);
  gemm_xwT_bias<true><<<gg, 256, 0, stream>>>(X16V, W16V, bv, VP, MROWS, DM, DM);

  size_t rope_total = (size_t)NB * LQ * NHD * 32;
  rope_reshape<<<nblk(rope_total), 256, 0, stream>>>(QP, QR, 0.125f);
  rope_reshape<<<nblk(rope_total), 256, 0, stream>>>(KP, KR, 1.0f);
  v_reshape<<<nblk(NLD), 256, 0, stream>>>(VP, VT);

  dim3 ga(NB * NHD, LQ / 128);
  flash_attn<<<ga, 256, 0, stream>>>(QR, KR, VT, AO);

  gemm_xwT_bias<false><<<gg, 256, 0, stream>>>(AO, W16O, bo, out, MROWS, DM, DM);
}
